// MultiHeadedAttention_455266533797
// MI455X (gfx1250) — matrix operand order repaired and checked by reading, not compiled, not run on hardware
//
#include <hip/hip_runtime.h>
#include <hip/hip_bf16.h>

typedef _Float16 half_t;
typedef __attribute__((ext_vector_type(16))) _Float16 v16h;
typedef __attribute__((ext_vector_type(8)))  _Float16 v8h;
typedef __attribute__((ext_vector_type(8)))  float    v8f;

#define BDIM 4
#define SDIM 2048
#define DMODEL 1024
#define HDIM 16
#define DHEAD 64

#define AS1 __attribute__((address_space(1)))

// reinterpret a generic pointer as an explicit global (AS1) pointer so the
// backend emits global_load/global_store instead of flat_* (FLAT ties up
// DScnt + LDS path; GLOBAL only uses LOADcnt/STOREcnt).
template <typename T>
__device__ __forceinline__ AS1 T* as_global(T* p) {
    return (AS1 T*)(unsigned long long)(uintptr_t)p;
}

// ---------------- fragment helpers (CDNA5 WMMA 16x16x32 f16 layouts) ----------

__device__ __forceinline__ v16h mk16(v8h lo, v8h hi) {
    v16h r;
#pragma unroll
    for (int i = 0; i < 8; ++i) { r[i] = lo[i]; r[i + 8] = hi[i]; }
    return r;
}

// A matrix 16x32 (MxK) from row-major [M][K]:
// lane<16 : row=lane,    halves = K k0..k0+15
// lane>=16: row=lane-16, halves = K k0+16..k0+31
// precomputed pointer p = A + row*lda + k0 + (lane>>4)*16 ; chunks at p and p+8
__device__ __forceinline__ v16h frag_a_g(const half_t* p) {
    return mk16(*as_global((const v8h*)p), *as_global((const v8h*)(p + 8)));
}

// B matrix 32x16 (KxN) from N-major [N][K]:
// lane<16 : col=lane,    halves = K k0..k0+15
// lane>=16: col=lane-16, halves = K k0+16..k0+31
// precomputed pointer p = W + col*ldb + k0 + (lane>>4)*16 ; chunks at p and p+8
__device__ __forceinline__ v16h frag_b_g(const half_t* p) {
    return mk16(*as_global((const v8h*)p), *as_global((const v8h*)(p + 8)));
}

// generic-addrspace A loader (used for LDS Pbuf -> ds_load path)
__device__ __forceinline__ v16h load_a_lds(const half_t* A, int lda, int m0, int k0, int lane) {
    int r = m0 + (lane & 15);
    const half_t* p = A + (size_t)r * lda + k0 + (lane >> 4) * 16;
    return mk16(*(const v8h*)p, *(const v8h*)(p + 8));
}

__device__ __forceinline__ v16h load_a_g(const half_t* A, int lda, int m0, int k0, int lane) {
    int r = m0 + (lane & 15);
    return frag_a_g(A + (size_t)r * lda + k0 + (lane >> 4) * 16);
}

#define WMMA_F16(a, b, c) __builtin_amdgcn_wmma_f32_16x16x32_f16(false, (a), false, (b), (short)0, (c), false, false)

// ---------------- f32 -> f16 conversion ---------------------------------------

__global__ void cvt_f32_f16(const float* __restrict__ in, half_t* __restrict__ out, int n) {
    int i = blockIdx.x * blockDim.x + threadIdx.x;
    int stride = gridDim.x * blockDim.x;
    for (; i < n; i += stride) as_global(out)[i] = (half_t)as_global(in)[i];
}

// ---------------- generic WMMA GEMM: C[M,N] = A[M,K] * W[N,K]^T + bias --------
// mode 0: out f16, scattered to Q/K layout [B,H,S,64]
// mode 1: out f16, scattered to Vt layout  [B,H,64,S] (transposed)
// mode 2: out f32, row-major [M,N]  (final projection -> d_out)
__global__ __launch_bounds__(256, 1)
void gemm_wmma(const half_t* __restrict__ A, const half_t* __restrict__ Wm,
               const float* __restrict__ bias, void* __restrict__ out,
               int M, int N, int K, int mode) {
    int lane  = threadIdx.x & 31;
    int wave  = threadIdx.x >> 5;
    int waveId = blockIdx.x * (blockDim.x >> 5) + wave;
    int mTiles = M >> 6;
    int tileM = (waveId % mTiles) * 64;
    int tileN = (waveId / mTiles) * 64;
    if (tileN >= N) return;

    int hh  = lane >> 4;
    int col = lane & 15;

    // hoisted per-lane fragment pointers, bumped by 32 halves per k-step
    const half_t* pA[4];
    const half_t* pB[4];
#pragma unroll
    for (int i = 0; i < 4; ++i)
        pA[i] = A + (size_t)(tileM + 16 * i + col) * K + hh * 16;
#pragma unroll
    for (int j = 0; j < 4; ++j)
        pB[j] = Wm + (size_t)(tileN + 16 * j + col) * K + hh * 16;

    v8f acc[4][4] = {};
    for (int k0 = 0; k0 < K; k0 += 32) {
        v16h af[4], bf[4];
#pragma unroll
        for (int i = 0; i < 4; ++i) { af[i] = frag_a_g(pA[i]); pA[i] += 32; }
#pragma unroll
        for (int j = 0; j < 4; ++j) { bf[j] = frag_b_g(pB[j]); pB[j] += 32; }
#pragma unroll
        for (int i = 0; i < 4; ++i)
#pragma unroll
            for (int j = 0; j < 4; ++j)
                acc[i][j] = WMMA_F16(af[i], bf[j], acc[i][j]);
    }

#pragma unroll
    for (int i = 0; i < 4; ++i) {
#pragma unroll
        for (int j = 0; j < 4; ++j) {
            int c = tileN + j * 16 + col;
            float bval = as_global(bias)[c];
            int rbase = tileM + i * 16 + hh * 8;
#pragma unroll
            for (int v = 0; v < 8; ++v) {
                int r = rbase + v;
                float val = acc[i][j][v] + bval;
                if (mode == 2) {
                    as_global((float*)out)[(size_t)r * N + c] = val;
                } else {
                    int b = r >> 11;          // r / S
                    int s = r & (SDIM - 1);   // r % S
                    int h = c >> 6;           // c / 64
                    int e = c & 63;           // c % 64
                    if (mode == 0)
                        as_global((half_t*)out)[(((size_t)(b * HDIM + h)) * SDIM + s) * DHEAD + e] = (half_t)val;
                    else
                        as_global((half_t*)out)[(((size_t)(b * HDIM + h)) * DHEAD + e) * SDIM + s] = (half_t)val;
                }
            }
        }
    }
}

// ---------------- flash attention (causal, scale 1/32 AFTER mask) -------------
// grid: (B*H) * (S/128) blocks, 128 threads (4 waves), 32 q-rows per wave.
__global__ __launch_bounds__(128, 1)
void flash_attn(const half_t* __restrict__ Qh, const half_t* __restrict__ Kh,
                const half_t* __restrict__ Vt, half_t* __restrict__ ctx) {
    __shared__ _Float16 Pbuf[4][32 * 32];

    int lane = threadIdx.x & 31;
    int wave = threadIdx.x >> 5;
    const int qblocks = SDIM / 128;
    int bh   = blockIdx.x / qblocks;
    int qblk = blockIdx.x % qblocks;
    int b = bh >> 4;             // bh / H
    int h = bh & (HDIM - 1);     // bh % H
    int qbase = qblk * 128 + wave * 32;

    const half_t* Qp = Qh + (size_t)bh * SDIM * DHEAD;
    const half_t* Kp = Kh + (size_t)bh * SDIM * DHEAD;
    const half_t* Vp = Vt + (size_t)bh * DHEAD * SDIM;

    int hh  = lane >> 4;
    int col = lane & 15;

    // preload Q fragments: 2 M-tiles x 2 K-steps over d_head=64
    v16h qf[2][2];
#pragma unroll
    for (int mi = 0; mi < 2; ++mi)
#pragma unroll
        for (int ks = 0; ks < 2; ++ks)
            qf[mi][ks] = load_a_g(Qp, DHEAD, qbase + mi * 16, ks * 32, lane);

    // hoisted streaming pointers
    // K fragment (B operand): col index = key t = kt + ni*16 + col, contraction over e
    const half_t* pK[2];
#pragma unroll
    for (int ni = 0; ni < 2; ++ni)
        pK[ni] = Kp + (size_t)(ni * 16 + col) * DHEAD + hh * 16;   // advance 32*DHEAD per kt
    // V fragment (B operand): col index = v-dim j*16+col, contraction over t
    const half_t* pV[4];
#pragma unroll
    for (int j = 0; j < 4; ++j)
        pV[j] = Vp + (size_t)(j * 16 + col) * SDIM + hh * 16;      // advance 32 per kt

    v8f o[2][4] = {};
    float mrow[2][8], lrow[2][8];
#pragma unroll
    for (int mi = 0; mi < 2; ++mi)
#pragma unroll
        for (int v = 0; v < 8; ++v) { mrow[mi][v] = -__builtin_inff(); lrow[mi][v] = 0.f; }

    const float scale = 0.03125f; // 1/sqrt(1024)

    int ktEnd = qbase + 32; // causal: keys t <= qbase+31
    for (int kt = 0; kt < ktEnd; kt += 32) {
        // scores: 32 q-rows x 32 keys
        v8f sacc[2][2] = {};
#pragma unroll
        for (int ks = 0; ks < 2; ++ks) {
            v16h kb0 = frag_b_g(pK[0] + ks * 32);
            v16h kb1 = frag_b_g(pK[1] + ks * 32);
#pragma unroll
            for (int mi = 0; mi < 2; ++mi) {
                sacc[mi][0] = WMMA_F16(qf[mi][ks], kb0, sacc[mi][0]);
                sacc[mi][1] = WMMA_F16(qf[mi][ks], kb1, sacc[mi][1]);
            }
        }
        pK[0] += 32 * DHEAD;
        pK[1] += 32 * DHEAD;

        // mask + scale
        float pv[2][2][8];
#pragma unroll
        for (int mi = 0; mi < 2; ++mi)
#pragma unroll
            for (int ni = 0; ni < 2; ++ni) {
                int t = kt + ni * 16 + col;
#pragma unroll
                for (int v = 0; v < 8; ++v) {
                    int q = qbase + mi * 16 + hh * 8 + v;
                    float val = sacc[mi][ni][v] * scale;
                    pv[mi][ni][v] = (t <= q) ? val : -__builtin_inff();
                }
            }
        // online softmax update
#pragma unroll
        for (int mi = 0; mi < 2; ++mi) {
#pragma unroll
            for (int v = 0; v < 8; ++v) {
                float x = fmaxf(pv[mi][0][v], pv[mi][1][v]);
                x = fmaxf(x, __shfl_xor(x, 1, 32));
                x = fmaxf(x, __shfl_xor(x, 2, 32));
                x = fmaxf(x, __shfl_xor(x, 4, 32));
                x = fmaxf(x, __shfl_xor(x, 8, 32));
                float mn   = fmaxf(mrow[mi][v], x);
                float corr = __expf(mrow[mi][v] - mn);
                mrow[mi][v] = mn;
                float rs = 0.f;
#pragma unroll
                for (int ni = 0; ni < 2; ++ni) {
                    float p0 = __expf(pv[mi][ni][v] - mn);
                    pv[mi][ni][v] = p0;
                    rs += p0;
                }
                rs += __shfl_xor(rs, 1, 32);
                rs += __shfl_xor(rs, 2, 32);
                rs += __shfl_xor(rs, 4, 32);
                rs += __shfl_xor(rs, 8, 32);
                lrow[mi][v] = lrow[mi][v] * corr + rs;
#pragma unroll
                for (int j = 0; j < 4; ++j) o[mi][j][v] *= corr;
            }
        }
        // C-layout -> A-fragment reshape through LDS
#pragma unroll
        for (int mi = 0; mi < 2; ++mi)
#pragma unroll
            for (int ni = 0; ni < 2; ++ni)
#pragma unroll
                for (int v = 0; v < 8; ++v) {
                    int rl = mi * 16 + hh * 8 + v;
                    int cl = ni * 16 + col;
                    Pbuf[wave][rl * 32 + cl] = (_Float16)pv[mi][ni][v];
                }
        asm volatile("s_wait_dscnt 0x0" ::: "memory");
        v16h pf[2];
#pragma unroll
        for (int mi = 0; mi < 2; ++mi)
            pf[mi] = load_a_lds(&Pbuf[wave][0], 32, mi * 16, 0, lane);
        // O += P * V   (Vt is [e][t], exactly the B operand layout)
#pragma unroll
        for (int j = 0; j < 4; ++j) {
            v16h vb = frag_b_g(pV[j]);
#pragma unroll
            for (int mi = 0; mi < 2; ++mi)
                o[mi][j] = WMMA_F16(pf[mi], vb, o[mi][j]);
        }
#pragma unroll
        for (int j = 0; j < 4; ++j) pV[j] += 32;
        asm volatile("s_wait_dscnt 0x0" ::: "memory");
    }

    // epilogue: normalize and store ctx f16 in [B*S, H*64] row-major
#pragma unroll
    for (int mi = 0; mi < 2; ++mi)
#pragma unroll
        for (int j = 0; j < 4; ++j)
#pragma unroll
            for (int v = 0; v < 8; ++v) {
                int q = qbase + mi * 16 + hh * 8 + v;
                int c = j * 16 + col;
                float val = o[mi][j][v] / lrow[mi][v];
                as_global(ctx)[((size_t)(b * SDIM + q)) * DMODEL + h * DHEAD + c] = (half_t)val;
            }
}

// ---------------- host launcher ----------------------------------------------

extern "C" void kernel_launch(void* const* d_in, const int* in_sizes, int n_in,
                              void* d_out, int out_size, void* d_ws, size_t ws_size,
                              hipStream_t stream) {
    const float* in_q = (const float*)d_in[0];
    const float* in_k = (const float*)d_in[1];
    const float* in_v = (const float*)d_in[2];
    const float* WQw  = (const float*)d_in[3];
    const float* WQb  = (const float*)d_in[4];
    const float* WKw  = (const float*)d_in[5];
    const float* WKb  = (const float*)d_in[6];
    const float* WVw  = (const float*)d_in[7];
    const float* WVb  = (const float*)d_in[8];
    const float* Ww   = (const float*)d_in[9];
    const float* Wb   = (const float*)d_in[10];

    const size_t M  = (size_t)BDIM * SDIM;     // 8192
    const size_t NX = M * DMODEL;              // 8,388,608  (inputs / ctx)
    const size_t NW = (size_t)DMODEL * DMODEL; // 1,048,576  (weights)
    const size_t NH = (size_t)BDIM * HDIM * SDIM * DHEAD; // 8,388,608

    half_t* ws = (half_t*)d_ws;
    size_t off = 0;
    half_t* Xq = ws + off; off += NX;
    half_t* Xk = ws + off; off += NX;
    half_t* Xv = ws + off; off += NX;
    half_t* Wq = ws + off; off += NW;
    half_t* Wk = ws + off; off += NW;
    half_t* Wv = ws + off; off += NW;
    half_t* Wo = ws + off; off += NW;
    half_t* Qh = ws + off; off += NH;
    half_t* Kh = ws + off; off += NH;
    half_t* Vt = ws + off; off += NH;
    half_t* Cx = ws + off; off += NX;

    // 1. precision conversion
    cvt_f32_f16<<<1024, 256, 0, stream>>>(in_q, Xq, (int)NX);
    cvt_f32_f16<<<1024, 256, 0, stream>>>(in_k, Xk, (int)NX);
    cvt_f32_f16<<<1024, 256, 0, stream>>>(in_v, Xv, (int)NX);
    cvt_f32_f16<<<512, 256, 0, stream>>>(WQw, Wq, (int)NW);
    cvt_f32_f16<<<512, 256, 0, stream>>>(WKw, Wk, (int)NW);
    cvt_f32_f16<<<512, 256, 0, stream>>>(WVw, Wv, (int)NW);
    cvt_f32_f16<<<512, 256, 0, stream>>>(Ww,  Wo, (int)NW);

    // 2. projections: (M/64)*(N/64)/8 waves-per-block blocks = 256 blocks of 256 thr
    int gblocks = (int)((M / 64) * (DMODEL / 64) / 8);
    gemm_wmma<<<gblocks, 256, 0, stream>>>(Xq, Wq, WQb, Qh, (int)M, DMODEL, DMODEL, 0);
    gemm_wmma<<<gblocks, 256, 0, stream>>>(Xk, Wk, WKb, Kh, (int)M, DMODEL, DMODEL, 0);
    gemm_wmma<<<gblocks, 256, 0, stream>>>(Xv, Wv, WVb, Vt, (int)M, DMODEL, DMODEL, 1);

    // 3. flash attention: B*H * (S/128) blocks, 128 threads
    flash_attn<<<BDIM * HDIM * (SDIM / 128), 128, 0, stream>>>(Qh, Kh, Vt, Cx);

    // 4. output projection -> fp32 d_out
    gemm_wmma<<<gblocks, 256, 0, stream>>>(Cx, Wo, Wb, d_out, (int)M, DMODEL, DMODEL, 2);
}
